// Decode_67860483276934
// MI455X (gfx1250) — matrix operand order repaired and checked by reading, not compiled, not run on hardware
//
#include <hip/hip_runtime.h>
#include <hip/hip_bf16.h>

typedef __bf16 bf16;
typedef __attribute__((ext_vector_type(16))) __bf16 v16bf;
typedef __attribute__((ext_vector_type(8)))  float  v8f;

#define HH 192
#define WW 192
#define HW (192*192)
#define CIN 64
#define CMID 256
#define NPTS 128
#define NCTR 128          // B*MAXC
#define KFC1 8256         // (NPT+1)*64
#define MAXC 32

static __device__ __forceinline__ v8f wmma_bf16(v16bf a, v16bf b, v8f c) {
    return __builtin_amdgcn_wmma_f32_16x16x32_bf16(false, a, false, b, (short)0, c, false, false);
}

// ---------------------------------------------------------------------------
// K0: repack conv weights into WMMA B-fragment layout (bf16).
// B1f[kt(18)][nt(16)][lane(32)][i(16)]  for GEMM1 (K=576, N=256)
// B2f[kt(8)][nt(4)][lane(32)][i(16)]    for GEMM2 (K=256, N=64)
// Fragment rule: col = lane&15, h = lane>>4, b[i] = B[kt*32 + (i>>3)*16 + h*8 + (i&7)][col]
// ---------------------------------------------------------------------------
__global__ __launch_bounds__(256) void repack_weights(const float* __restrict__ w1,
                                                      const float* __restrict__ w2,
                                                      bf16* __restrict__ B1f,
                                                      bf16* __restrict__ B2f) {
    int id = blockIdx.x * 256 + threadIdx.x;
    const int N1 = 18 * 16 * 32 * 16;   // 147456
    if (id < N1) {
        int i    = id & 15;
        int lane = (id >> 4) & 31;
        int nt   = (id >> 9) & 15;
        int kt   = id >> 13;
        int k = kt * 32 + ((i >> 3) << 4) + ((lane >> 4) << 3) + (i & 7);       // 0..575
        int n = nt * 16 + (lane & 15);                  // 0..255
        int tap = k >> 6, c = k & 63;
        int dy = tap / 3, dx = tap % 3;
        float v = w1[((n * 64 + c) * 3 + dy) * 3 + dx]; // conv1_w[n][c][dy][dx]
        B1f[id] = (bf16)v;
    }
    int id2 = id - N1;
    if (id2 >= 0 && id2 < 8 * 4 * 32 * 16) {
        int i    = id2 & 15;
        int lane = (id2 >> 4) & 31;
        int nt   = (id2 >> 9) & 3;
        int kt   = id2 >> 11;
        int k = kt * 32 + ((i >> 3) << 4) + ((lane >> 4) << 3) + (i & 7);       // 0..255
        int n = nt * 16 + (lane & 15);                  // 0..63
        B2f[id2] = (bf16)w2[n * 256 + k];               // conv2_w[n][k][0][0]
    }
}

// ---------------------------------------------------------------------------
// K1: gather init_polys from wh_pred; write init (ws) and init*DOWN (out0)
// ---------------------------------------------------------------------------
__global__ __launch_bounds__(256) void gather_init(const float* __restrict__ wh,
                                                   const int* __restrict__ ct_ind,
                                                   const int* __restrict__ ct_img,
                                                   float* __restrict__ initp,
                                                   float* __restrict__ out0) {
    int id = blockIdx.x * 256 + threadIdx.x;
    if (id >= NCTR * NPTS) return;
    int n = id >> 7, j = id & 127;
    int ind = ct_ind[n];
    int img = ct_img[n];
    int cx = ind % WW, cy = ind / WW;
    int base = img * 2 * NPTS * HW + cy * WW + cx;
    float ox = wh[base + (2 * j) * HW];
    float oy = wh[base + (2 * j + 1) * HW];
    float ix = ox * 10.0f + (float)cx;
    float iy = oy * 10.0f + (float)cy;
    initp[id * 2]     = ix;
    initp[id * 2 + 1] = iy;
    out0[id * 2]      = ix * 4.0f;   // DOWN
    out0[id * 2 + 1]  = iy * 4.0f;
}

// ---------------------------------------------------------------------------
// K2: fused conv1(3x3,64->256)+bias+ReLU -> conv2(1x1,256->64)+bias, WMMA bf16
// One workgroup = 64 contiguous pixels of one image row. 8 waves.
// Both kt loops fully unrolled: A LDS addresses become p*128 + hi*16 + const
// (pure immediate-offset ds_load_b128), B addresses fold into the signed
// 24-bit immediate of global_load_b128 on a single base VGPR.
// ---------------------------------------------------------------------------
__global__ __launch_bounds__(256) void conv_fused(const float* __restrict__ x,
                                                  const float* __restrict__ b1,
                                                  const float* __restrict__ b2,
                                                  const bf16* __restrict__ B1f,
                                                  const bf16* __restrict__ B2f,
                                                  float* __restrict__ feat) {
    __shared__ bf16 xp[3 * 66 * 64];    // [dy][xx][c], halo patch
    __shared__ bf16 tl[64 * 256];       // [pixel][mid-channel]

    int wg = blockIdx.x;                // b*H*3 + y*3 + xt
    int xt = wg % 3;
    int y  = (wg / 3) % HH;
    int b  = wg / (3 * HH);
    int x0 = xt * 64;

    // stage input halo patch, fp32 -> bf16
    for (int idx = threadIdx.x; idx < 3 * 66 * 64; idx += 256) {
        int xxi = idx % 66;
        int c   = (idx / 66) & 63;
        int dy  = idx / (66 * 64);
        int gy = y + dy - 1;
        int gx = x0 + xxi - 1;
        float v = 0.0f;
        if (gy >= 0 && gy < HH && gx >= 0 && gx < WW)
            v = x[((b * CIN + c) * HH + gy) * WW + gx];
        xp[(dy * 66 + xxi) * 64 + c] = (bf16)v;
    }
    __syncthreads();

    int lane  = threadIdx.x & 31;
    int wave  = threadIdx.x >> 5;
    int row   = lane & 15;
    int hi    = lane >> 4;              // 0/1
    int mtile = wave & 3;               // 4 M-tiles of 16 pixels
    int ntg   = (wave >> 2) * 8;        // 8 N-tiles per wave (of 16)
    int p     = mtile * 16 + row;       // this lane's A row (pixel)

    // ---------------- GEMM1: t = relu(conv1(x)+b1), K = 576 -----------------
    v8f acc[8];
#pragma unroll
    for (int t = 0; t < 8; ++t) acc[t] = (v8f){0,0,0,0,0,0,0,0};

    const bf16* b1base = &B1f[(ntg * 32 + lane) * 16];
#pragma unroll
    for (int kt = 0; kt < 18; ++kt) {
        // A fragment: both 8-element halves are contiguous runs in xp
        v16bf a;
        int kb = kt * 32 + hi * 8;
#pragma unroll
        for (int half = 0; half < 2; ++half) {
#pragma unroll
            for (int j = 0; j < 4; ++j) {
                int k = kb + half * 16 + j * 2;        // even, pair stays in-tap
                int tap = k >> 6, c = k & 63;
                int dy = tap / 3, dx = tap % 3;
                const bf16* s = &xp[(dy * 66 + (p + dx)) * 64 + c];
                a[half * 8 + j * 2]     = s[0];
                a[half * 8 + j * 2 + 1] = s[1];
            }
        }
        // batch all 8 B fragments (constant offsets from one base)
        v16bf bb[8];
#pragma unroll
        for (int t = 0; t < 8; ++t)
            bb[t] = *(const v16bf*)(b1base + (kt * 16 + t) * (32 * 16));
#pragma unroll
        for (int t = 0; t < 8; ++t)
            acc[t] = wmma_bf16(a, bb[t], acc[t]);
    }

    // epilogue: bias + relu + bf16 -> tl[p][n]
#pragma unroll
    for (int t = 0; t < 8; ++t) {
        int n = (ntg + t) * 16 + row;
        float bias = b1[n];
#pragma unroll
        for (int r = 0; r < 8; ++r) {
            float v = acc[t][r] + bias;
            v = v > 0.0f ? v : 0.0f;
            int pp = mtile * 16 + r + hi * 8;
            tl[pp * 256 + n] = (bf16)v;
        }
    }
    __syncthreads();

    // ---------------- GEMM2: feat = conv2(t)+b2, K = 256 --------------------
    int ntb = (wave >> 2) * 2;          // waves 0-3 -> nt {0,1}, 4-7 -> {2,3}
    v8f acc2[2];
    acc2[0] = (v8f){0,0,0,0,0,0,0,0};
    acc2[1] = (v8f){0,0,0,0,0,0,0,0};

    const bf16* b2base = &B2f[(ntb * 32 + lane) * 16];
#pragma unroll
    for (int kt = 0; kt < 8; ++kt) {
        v16bf a;
        int kb = kt * 32 + hi * 8;
#pragma unroll
        for (int half = 0; half < 2; ++half) {
#pragma unroll
            for (int j = 0; j < 4; ++j) {
                int k = kb + half * 16 + j * 2;
                const bf16* s = &tl[p * 256 + k];
                a[half * 8 + j * 2]     = s[0];
                a[half * 8 + j * 2 + 1] = s[1];
            }
        }
        v16bf bb[2];
#pragma unroll
        for (int t = 0; t < 2; ++t)
            bb[t] = *(const v16bf*)(b2base + (kt * 4 + t) * (32 * 16));
#pragma unroll
        for (int t = 0; t < 2; ++t)
            acc2[t] = wmma_bf16(a, bb[t], acc2[t]);
    }

#pragma unroll
    for (int t = 0; t < 2; ++t) {
        int n = (ntb + t) * 16 + row;   // out channel 0..63
        float bias = b2[n];
#pragma unroll
        for (int r = 0; r < 8; ++r) {
            int pp = mtile * 16 + r + hi * 8;
            feat[((b * CIN + n) * HH + y) * WW + x0 + pp] = acc2[t][r] + bias;
        }
    }
}

// ---------------------------------------------------------------------------
// K3/K8: point-in-polygon mask -> mb[b][c][H][W] (reference _mask_batch packing)
// ---------------------------------------------------------------------------
__global__ __launch_bounds__(256) void pnp_kernel(const float* __restrict__ polys,
                                                  const int* __restrict__ ct_num,
                                                  float* __restrict__ mb) {
    __shared__ float ex1[NPTS], ey1[NPTS], edx[NPTS], ey2[NPTS], einv[NPTS];
    int bc = blockIdx.y;                 // b*MAXC + c
    int bb = bc >> 5, c = bc & 31;
    int cnb = ct_num[bb];
    int start = (bb == 0) ? 0 : ct_num[bb - 1];   // matches reference _mask_batch
    bool active = (c < cnb);
    if (active) {
        const float* P = polys + (start + c) * NPTS * 2;
        for (int j = threadIdx.x; j < NPTS; j += 256) {
            float x1 = P[j * 2], y1 = P[j * 2 + 1];
            int j2 = (j + 1) & (NPTS - 1);
            float x2 = P[j2 * 2], y2 = P[j2 * 2 + 1];
            float den = y2 - y1;
            if (fabsf(den) < 1e-9f) den = 1e-9f;
            ex1[j] = x1; ey1[j] = y1; edx[j] = x2 - x1; ey2[j] = y2;
            einv[j] = 1.0f / den;
        }
    }
    __syncthreads();
    int pix = blockIdx.x * 256 + threadIdx.x;
    if (pix >= HW) return;
    float yy = (float)(pix / WW);
    float xx = (float)(pix % WW);
    bool par = false;
    if (active) {
        for (int j = 0; j < NPTS; ++j) {
            bool cond = (ey1[j] > yy) != (ey2[j] > yy);
            float xint = ex1[j] + edx[j] * (yy - ey1[j]) * einv[j];
            par ^= (cond && (xx < xint));
        }
    }
    mb[bc * HW + pix] = par ? 1.0f : 0.0f;
}

// ---------------------------------------------------------------------------
// K4/K9: feat_out = relu(feat_in * (1 + max_c mb[b][c]))
// ---------------------------------------------------------------------------
__global__ __launch_bounds__(256) void maskfuse(const float* __restrict__ mb,
                                                const float* __restrict__ fin,
                                                float* __restrict__ fout) {
    int id = blockIdx.x * 256 + threadIdx.x;
    if (id >= 4 * HW) return;
    int b = id / HW, pix = id % HW;
    const float* mp = mb + b * MAXC * HW + pix;
    float m = 0.0f;
#pragma unroll 4
    for (int c = 0; c < MAXC; ++c) m = fmaxf(m, mp[c * HW]);
    float s = 1.0f + m;
    const float* fi = fin + b * CIN * HW + pix;
    float* fo = fout + b * CIN * HW + pix;
#pragma unroll 4
    for (int c = 0; c < CIN; ++c) {
        float v = fi[c * HW] * s;
        fo[c * HW] = v > 0.0f ? v : 0.0f;
    }
}

// ---------------------------------------------------------------------------
// K5: bilinear sample feat at [center; init_polys] -> fp[n][c*129 + p]
// ---------------------------------------------------------------------------
__global__ __launch_bounds__(256) void sample_kernel(const float* __restrict__ feat,
                                                     const float* __restrict__ initp,
                                                     const int* __restrict__ ct_ind,
                                                     const int* __restrict__ ct_img,
                                                     float* __restrict__ fp) {
    int id = blockIdx.x * 256 + threadIdx.x;
    if (id >= NCTR * 129) return;
    int n = id / 129, p = id % 129;
    int img = ct_img[n];
    float ptx, pty;
    if (p == 0) {
        int ind = ct_ind[n];
        ptx = (float)(ind % WW);
        pty = (float)(ind / WW);
    } else {
        ptx = initp[(n * NPTS + (p - 1)) * 2];
        pty = initp[(n * NPTS + (p - 1)) * 2 + 1];
    }
    float xs = ptx - 0.5f, ys = pty - 0.5f;
    float x0f = floorf(xs), y0f = floorf(ys);
    float wx = xs - x0f, wy = ys - y0f;
    int x0 = (int)x0f, y0 = (int)y0f;
    int xi[2] = {x0, x0 + 1}, yi[2] = {y0, y0 + 1};
    float w[4] = {(1 - wx) * (1 - wy), wx * (1 - wy), (1 - wx) * wy, wx * wy};
    int   ofs[4];
    float wv[4];
#pragma unroll
    for (int q = 0; q < 4; ++q) {
        int xq = xi[q & 1], yq = yi[q >> 1];
        bool valid = (xq >= 0) && (xq < WW) && (yq >= 0) && (yq < HH);
        int xc = xq < 0 ? 0 : (xq > WW - 1 ? WW - 1 : xq);
        int yc = yq < 0 ? 0 : (yq > HH - 1 ? HH - 1 : yq);
        ofs[q] = yc * WW + xc;
        wv[q]  = valid ? w[q] : 0.0f;
    }
    const float* fb = feat + img * CIN * HW;
    for (int c = 0; c < CIN; ++c) {
        const float* fc = fb + c * HW;
        float v = fc[ofs[0]] * wv[0] + fc[ofs[1]] * wv[1] +
                  fc[ofs[2]] * wv[2] + fc[ofs[3]] * wv[3];
        fp[n * KFC1 + c * 129 + p] = v;
    }
}

// ---------------------------------------------------------------------------
// FC fragment loader (fp32 -> bf16 on the fly)
// ---------------------------------------------------------------------------
static __device__ __forceinline__ void fc_load_frag(const float* __restrict__ A,
                                                    const float* __restrict__ Wt,
                                                    int m, int n, int K, int kt,
                                                    int hi, v16bf* a, v16bf* bb) {
    int kb = kt * 32 + hi * 16;
#pragma unroll
    for (int half = 0; half < 2; ++half) {
        const float* ap = A + m * K + kb + half * 8;
#pragma unroll
        for (int j = 0; j < 8; ++j) (*a)[half * 8 + j] = (bf16)ap[j];
    }
    const float* bp = Wt + n * K + kt * 32 + hi * 16;
#pragma unroll
    for (int j = 0; j < 16; ++j) (*bb)[j] = (bf16)bp[j];
}

// ---------------------------------------------------------------------------
// K6a: h(128x512) = fp(128x8256) @ w_poly^T   (WMMA bf16)
// grid.x = N/16 = 32, 8 waves = 8 M-tiles.  kt unrolled x4: four load
// groups in flight before the first WMMA waits.
// ---------------------------------------------------------------------------
__global__ __launch_bounds__(256) void fc1_kernel(const float* __restrict__ A,
                                                  const float* __restrict__ Wt,
                                                  float* __restrict__ Hout) {
    int lane = threadIdx.x & 31;
    int wave = threadIdx.x >> 5;
    int row = lane & 15, hi = lane >> 4;
    int m = wave * 16 + row;
    int nt = blockIdx.x;
    int n = nt * 16 + row;
    v8f acc = (v8f){0,0,0,0,0,0,0,0};
    for (int kt = 0; kt < KFC1 / 32; kt += 4) {        // 258 steps total
        v16bf a0, b0, a1, b1, a2, b2, a3, b3;
        fc_load_frag(A, Wt, m, n, KFC1, kt,     hi, &a0, &b0);
        fc_load_frag(A, Wt, m, n, KFC1, kt + 1, hi, &a1, &b1);
        fc_load_frag(A, Wt, m, n, KFC1, kt + 2, hi, &a2, &b2);
        fc_load_frag(A, Wt, m, n, KFC1, kt + 3, hi, &a3, &b3);
        acc = wmma_bf16(a0, b0, acc);
        acc = wmma_bf16(a1, b1, acc);
        acc = wmma_bf16(a2, b2, acc);
        acc = wmma_bf16(a3, b3, acc);
    }
    // KFC1/32 = 258 is not divisible by 4: handle the tail (kt = 256, 257)
    {
        v16bf a0, b0, a1, b1;
        fc_load_frag(A, Wt, m, n, KFC1, 256, hi, &a0, &b0);
        fc_load_frag(A, Wt, m, n, KFC1, 257, hi, &a1, &b1);
        acc = wmma_bf16(a0, b0, acc);
        acc = wmma_bf16(a1, b1, acc);
    }
#pragma unroll
    for (int r = 0; r < 8; ++r)
        Hout[(wave * 16 + r + hi * 8) * 512 + nt * 16 + row] = acc[r];
}

// ---------------------------------------------------------------------------
// K6b: off = h(128x512) @ w_fuse^T + b_fuse;  coarse = off*4 + init
// writes ws.coarse and out1 = coarse*DOWN.  grid.x = 256/16 = 16.
// ---------------------------------------------------------------------------
__global__ __launch_bounds__(256) void fc2_kernel(const float* __restrict__ A,
                                                  const float* __restrict__ Wt,
                                                  const float* __restrict__ bfuse,
                                                  const float* __restrict__ initp,
                                                  float* __restrict__ coarse,
                                                  float* __restrict__ out1) {
    int lane = threadIdx.x & 31;
    int wave = threadIdx.x >> 5;
    int row = lane & 15, hi = lane >> 4;
    int m = wave * 16 + row;
    int nt = blockIdx.x;
    int col = nt * 16 + row;            // output index j in 0..255
    v8f acc = (v8f){0,0,0,0,0,0,0,0};
#pragma unroll
    for (int kt = 0; kt < 16; kt += 2) {
        v16bf a0, b0, a1, b1;
        fc_load_frag(A, Wt, m, col, 512, kt,     hi, &a0, &b0);
        fc_load_frag(A, Wt, m, col, 512, kt + 1, hi, &a1, &b1);
        acc = wmma_bf16(a0, b0, acc);
        acc = wmma_bf16(a1, b1, acc);
    }
    float bias = bfuse[col];
    int pt = col >> 1, dd = col & 1;
#pragma unroll
    for (int r = 0; r < 8; ++r) {
        int mr = wave * 16 + r + hi * 8;            // center index
        float d = acc[r] + bias;
        float co = d * 4.0f + initp[(mr * NPTS + pt) * 2 + dd];   // COARSE_STRIDE
        coarse[(mr * NPTS + pt) * 2 + dd] = co;
        out1[(mr * NPTS + pt) * 2 + dd]   = co * 4.0f;            // DOWN
    }
}

// ---------------------------------------------------------------------------
extern "C" void kernel_launch(void* const* d_in, const int* in_sizes, int n_in,
                              void* d_out, int out_size, void* d_ws, size_t ws_size,
                              hipStream_t stream) {
    const float* wh_pred  = (const float*)d_in[0];
    const float* cnn_feat = (const float*)d_in[1];
    const float* conv1_w  = (const float*)d_in[2];
    const float* conv1_b  = (const float*)d_in[3];
    const float* conv2_w  = (const float*)d_in[4];
    const float* conv2_b  = (const float*)d_in[5];
    const float* w_poly   = (const float*)d_in[6];
    const float* w_fuse   = (const float*)d_in[7];
    const float* b_fuse   = (const float*)d_in[8];
    const int*   ct_ind   = (const int*)d_in[9];
    const int*   ct_img   = (const int*)d_in[10];
    const int*   ct_num   = (const int*)d_in[11];

    float* out = (float*)d_out;
    // output regions (flat concat in return order)
    float* out_init  = out;                       // 128*128*2
    float* out_coar  = out + 32768;               // 128*128*2
    float* out_mbi   = out + 65536;               // 4*32*HW
    float* out_mbc   = out + 65536 + 4 * MAXC * HW;
    float* out_feat  = out + 65536 + 8 * MAXC * HW;

    // workspace layout (bytes)
    char* ws = (char*)d_ws;
    float* ws_init   = (float*)(ws + 0);          // 131072 B
    float* ws_coarse = (float*)(ws + 131072);     // 131072 B
    bf16*  ws_B1f    = (bf16*)(ws + 262144);      // 294912 B
    bf16*  ws_B2f    = (bf16*)(ws + 557056);      // 32768 B
    float* ws_feat   = (float*)(ws + 589824);     // 37748736 B
    float* ws_fp     = (float*)(ws + 38338560);   // 4227072 B
    float* ws_h      = (float*)(ws + 42565632);   // 262144 B

    // 0) weight repack to WMMA fragments
    repack_weights<<<(147456 + 16384 + 255) / 256, 256, 0, stream>>>(
        conv1_w, conv2_w, ws_B1f, ws_B2f);

    // 1) init polys
    gather_init<<<(NCTR * NPTS + 255) / 256, 256, 0, stream>>>(
        wh_pred, ct_ind, ct_img, ws_init, out_init);

    // 2) fused conv (WMMA): feat0 -> ws_feat
    conv_fused<<<4 * HH * 3, 256, 0, stream>>>(
        cnn_feat, conv1_b, conv2_b, ws_B1f, ws_B2f, ws_feat);

    // 3) pnp(init) -> mb_init
    dim3 pgrid((HW + 255) / 256, 4 * MAXC);
    pnp_kernel<<<pgrid, 256, 0, stream>>>(ws_init, ct_num, out_mbi);

    // 4) feat1 = relu(feat0*(1+max mb_init))  (in place in ws_feat)
    maskfuse<<<(4 * HW + 255) / 256, 256, 0, stream>>>(out_mbi, ws_feat, ws_feat);

    // 5) bilinear sample -> ws_fp
    sample_kernel<<<(NCTR * 129 + 255) / 256, 256, 0, stream>>>(
        ws_feat, ws_init, ct_ind, ct_img, ws_fp);

    // 6) FC chain (WMMA)
    fc1_kernel<<<32, 256, 0, stream>>>(ws_fp, w_poly, ws_h);
    fc2_kernel<<<16, 256, 0, stream>>>(ws_h, w_fuse, b_fuse, ws_init,
                                       ws_coarse, out_coar);

    // 7) pnp(coarse) -> mb_c
    pnp_kernel<<<pgrid, 256, 0, stream>>>(ws_coarse, ct_num, out_mbc);

    // 8) final feat = relu(feat1*(1+max mb_c)) -> out
    maskfuse<<<(4 * HW + 255) / 256, 256, 0, stream>>>(out_mbc, ws_feat, out_feat);
}
